// MixLoraFFN_59717225284242
// MI455X (gfx1250) — matrix operand order repaired and checked by reading, not compiled, not run on hardware
//
#include <hip/hip_runtime.h>
#include <math.h>
#include <stdint.h>

typedef __attribute__((ext_vector_type(16))) __bf16 v16bf;
typedef __attribute__((ext_vector_type(8)))  float  v8f;
typedef __attribute__((ext_vector_type(2)))  float  v2f;
typedef __attribute__((ext_vector_type(4)))  unsigned int v4u;
typedef __attribute__((ext_vector_type(8)))  int    v8i;
typedef __attribute__((ext_vector_type(4)))  int    v4i;

constexpr int   T_TOK  = 4096;   // B*S
constexpr int   H_DIM  = 2048;
constexpr int   I_DIM  = 8192;
constexpr int   NE     = 8;
constexpr int   RNK    = 16;
constexpr float SCALING = 2.0f;  // 32/16

constexpr int   SLAB   = 512;               // TDM slab width (elems)
constexpr int   SLAB_E = 16 * SLAB;         // elems per slab buffer

// ---------- helpers ----------
__device__ __forceinline__ unsigned short f2bf(float f) {
  unsigned int u = __float_as_uint(f);
  u += 0x7fffu + ((u >> 16) & 1u);          // round-to-nearest-even
  return (unsigned short)(u >> 16);
}

union AFrag  { v16bf v; uint4 q[2]; };            // 16 bf16 = 32B
union BFragW { v16bf v; unsigned short s[16]; };  // build bf16 from fp32
union C8     { v8f v; float f[8]; };

__device__ __forceinline__ v8f wmma_bf16(v16bf a, v16bf b, v8f c) {
  return __builtin_amdgcn_wmma_f32_16x16x32_bf16(false, a, false, b, (short)0, c, false, false);
}
__device__ __forceinline__ v8f wmma_f32(v2f a, v2f b, v8f c) {
  return __builtin_amdgcn_wmma_f32_16x16x4_f32(false, a, false, b, (short)0, c, false, false);
}

// ---------- TDM: 2D bf16 tile Global -> LDS (D# per CDNA5 ISA ch.8) ----------
// Issued by one wave; tracked with TENSORcnt.
__device__ __forceinline__ void tdm_load_2d_bf16(unsigned int lds_byte_off,
                                                 const unsigned short* gptr,
                                                 unsigned int cols,   // tile_dim0, elems
                                                 unsigned int rows,   // tile_dim1
                                                 unsigned long long row_stride) { // elems
  unsigned long long ga = (unsigned long long)(uintptr_t)gptr;
  v4u g0;
  g0[0] = 1u;                                            // count=1, user load
  g0[1] = lds_byte_off;                                  // lds_addr
  g0[2] = (unsigned int)(ga & 0xffffffffu);              // global_addr[31:0]
  g0[3] = (unsigned int)((ga >> 32) & 0x1ffffffu)        // global_addr[56:32]
          | (2u << 30);                                  // type=2 ("image")
  v8i g1;
  g1[0] = (int)(1u << 16);                               // wg_mask=0, data_size=1 (2B)
  g1[1] = (int)((cols & 0xffffu) << 16);                 // tensor_dim0[15:0]
  g1[2] = (int)(((cols >> 16) & 0xffffu) | ((rows & 0xffffu) << 16)); // dim0[31:16]|dim1[15:0]
  g1[3] = (int)(((rows >> 16) & 0xffffu) | ((cols & 0xffffu) << 16)); // dim1[31:16]|tile_dim0
  g1[4] = (int)(rows & 0xffffu);                         // tile_dim1, tile_dim2=0
  g1[5] = (int)(row_stride & 0xffffffffu);               // dim0_stride[31:0]
  g1[6] = (int)((row_stride >> 32) & 0xffffu);           // dim0_stride[47:32], dim1_stride=0
  g1[7] = 0;
  v4i z4 = {0, 0, 0, 0};
#if __clang_major__ >= 23
  v8i z8 = {0, 0, 0, 0, 0, 0, 0, 0};
  __builtin_amdgcn_tensor_load_to_lds(g0, g1, z4, z4, z8, 0);
#else
  __builtin_amdgcn_tensor_load_to_lds(g0, g1, z4, z4, 0);
#endif
}

// ---------- 1) fp32 -> bf16 convert ----------
__global__ void k_cvt(const float* __restrict__ src, unsigned short* __restrict__ dst, long n) {
  long i = (long)blockIdx.x * blockDim.x + threadIdx.x;
  long stride = (long)gridDim.x * blockDim.x;
  for (; i < n; i += stride) dst[i] = f2bf(src[i]);
}

// ---------- 2) router: wave per token ----------
__global__ void k_router(const float* __restrict__ xf, const float* __restrict__ rW,
                         float* __restrict__ w_dense, float* __restrict__ probs_sum,
                         float* __restrict__ sel_cnt) {
  int t = blockIdx.x;
  int lane = threadIdx.x;
  float logits[NE];
  const float* xr = xf + (long)t * H_DIM;
  for (int e = 0; e < NE; ++e) {
    const float* wr = rW + (long)e * H_DIM;
    float acc = 0.f;
    for (int h = lane; h < H_DIM; h += 32) acc += xr[h] * wr[h];
    for (int off = 16; off > 0; off >>= 1) acc += __shfl_xor(acc, off, 32);
    logits[e] = acc;
  }
  float mx = logits[0];
  for (int e = 1; e < NE; ++e) mx = fmaxf(mx, logits[e]);
  float p[NE], s = 0.f;
  for (int e = 0; e < NE; ++e) { p[e] = __expf(logits[e] - mx); s += p[e]; }
  float inv = 1.f / s;
  for (int e = 0; e < NE; ++e) p[e] *= inv;
  int i1 = 0;
  for (int e = 1; e < NE; ++e) if (p[e] > p[i1]) i1 = e;
  int i2 = (i1 == 0) ? 1 : 0;
  for (int e = 0; e < NE; ++e) if (e != i1 && p[e] > p[i2]) i2 = e;
  float wsum = p[i1] + p[i2];
  if (lane == 0) {
    for (int e = 0; e < NE; ++e) {
      float w = (e == i1) ? p[i1] / wsum : ((e == i2) ? p[i2] / wsum : 0.f);
      w_dense[t * NE + e] = w;
      atomicAdd(&probs_sum[e], p[e]);
    }
    atomicAdd(&sel_cnt[i1], 1.f);
    atomicAdd(&sel_cnt[i2], 1.f);
  }
}

// ---------- 3) LoRA A projections: ta = x @ A_e^T  [E,T,16] ----------
__global__ void k_loraA(const unsigned short* __restrict__ xb,
                        const float* __restrict__ gA, const float* __restrict__ uA,
                        float* __restrict__ taG, float* __restrict__ taU) {
  int tt = blockIdx.x, e = blockIdx.y;
  int lane = threadIdx.x;
  int t0 = tt * 16;
  int n   = lane & 15;
  int abk = (lane < 16) ? 0 : 16;
  int bbk = (lane < 16) ? 0 : 16;
  int mh  = (lane < 16) ? 0 : 8;
  v8f cg = {}, cu = {};
  const unsigned short* xrow = xb + (long)(t0 + n) * H_DIM;
  const float* gArow = gA + ((long)e * RNK + n) * H_DIM;
  const float* uArow = uA + ((long)e * RNK + n) * H_DIM;
  for (int h0 = 0; h0 < H_DIM; h0 += 32) {
    AFrag a;
    a.q[0] = *reinterpret_cast<const uint4*>(xrow + h0 + abk);
    a.q[1] = *reinterpret_cast<const uint4*>(xrow + h0 + abk + 8);
    BFragW bg, bu;
    const float4* g4 = reinterpret_cast<const float4*>(gArow + h0 + bbk);
    const float4* u4 = reinterpret_cast<const float4*>(uArow + h0 + bbk);
#pragma unroll
    for (int q = 0; q < 4; ++q) {
      float4 gv = g4[q], uv = u4[q];
      bg.s[4*q+0]=f2bf(gv.x); bg.s[4*q+1]=f2bf(gv.y); bg.s[4*q+2]=f2bf(gv.z); bg.s[4*q+3]=f2bf(gv.w);
      bu.s[4*q+0]=f2bf(uv.x); bu.s[4*q+1]=f2bf(uv.y); bu.s[4*q+2]=f2bf(uv.z); bu.s[4*q+3]=f2bf(uv.w);
    }
    cg = wmma_bf16(a.v, bg.v, cg);
    cu = wmma_bf16(a.v, bu.v, cu);
  }
  C8 og, ou; og.v = cg; ou.v = cu;
  for (int j = 0; j < 8; ++j) {
    int t = t0 + j + mh;
    taG[((long)e * T_TOK + t) * RNK + n] = og.f[j];
    taU[((long)e * T_TOK + t) * RNK + n] = ou.f[j];
  }
}

// ---------- 4) main: base gate/up + LoRA + SwiGLU + weighted combine + rw ----------
// Block: 4 waves, 16 tokens x 256 I. x tile double-buffered TDM pipeline (512-col slabs).
__global__ __launch_bounds__(128) void k_main(
                       const unsigned short* __restrict__ xb,
                       const unsigned short* __restrict__ gWb,
                       const unsigned short* __restrict__ uWb,
                       const float* __restrict__ taG, const float* __restrict__ taU,
                       const float* __restrict__ gB,  const float* __restrict__ uB,
                       const float* __restrict__ dA,
                       const float* __restrict__ w_dense,
                       unsigned short* __restrict__ aib,
                       float* __restrict__ rw) {
  __shared__ unsigned short xlds[2 * SLAB_E];  // 2 x 16KB TDM ping-pong buffers
  __shared__ float ldsWi[4 * 256];             // per-wave transpose scratch
  int tt = blockIdx.x, ib = blockIdx.y;
  int tid  = threadIdx.x;
  int wave = tid >> 5;
  int lane = tid & 31;
  int t0 = tt * 16;
  int n   = lane & 15;
  int m   = n;
  int abk = (lane < 16) ? 0 : 16;
  int bbk = (lane < 16) ? 0 : 16;
  int kk  = (lane < 16) ? 0 : 2;
  int mh  = (lane < 16) ? 0 : 8;

  v8f zero = {};
  v8f gacc[4], uacc[4];
#pragma unroll
  for (int it = 0; it < 4; ++it) { gacc[it] = zero; uacc[it] = zero; }
  v8f rwv[NE];
  for (int e = 0; e < NE; ++e) rwv[e] = zero;

  unsigned int ldsBase = (unsigned int)(uintptr_t)xlds;
  constexpr int NSLAB = H_DIM / SLAB;          // 4

  // ---- base gate/up projections over K = H; TDM prefetch pipelined with WMMA ----
  if (wave == 0)
    tdm_load_2d_bf16(ldsBase, xb + (long)t0 * H_DIM, SLAB, 16u,
                     (unsigned long long)H_DIM);
  for (int hc = 0; hc < NSLAB; ++hc) {
    if (wave == 0) __builtin_amdgcn_s_wait_tensorcnt(0);   // slab hc landed
    __syncthreads();
    if (wave == 0 && hc + 1 < NSLAB)                       // overlap DMA of hc+1
      tdm_load_2d_bf16(ldsBase + ((hc + 1) & 1) * (SLAB_E * 2),
                       xb + (long)t0 * H_DIM + (hc + 1) * SLAB, SLAB, 16u,
                       (unsigned long long)H_DIM);
    const unsigned short* xrow = xlds + (hc & 1) * SLAB_E + m * SLAB;
    for (int h0 = 0; h0 < SLAB; h0 += 32) {
      AFrag a;
      a.q[0] = *reinterpret_cast<const uint4*>(xrow + h0 + abk);       // ds_load_b128
      a.q[1] = *reinterpret_cast<const uint4*>(xrow + h0 + abk + 8);
      int hg = hc * SLAB + h0;
#pragma unroll
      for (int it = 0; it < 4; ++it) {
        int i0 = ib * 256 + (it * 4 + wave) * 16;
        const unsigned short* gwrow = gWb + (long)(i0 + n) * H_DIM + hg;
        const unsigned short* uwrow = uWb + (long)(i0 + n) * H_DIM + hg;
        AFrag bg, bu;
        bg.q[0] = *reinterpret_cast<const uint4*>(gwrow + bbk);
        bg.q[1] = *reinterpret_cast<const uint4*>(gwrow + bbk + 8);
        bu.q[0] = *reinterpret_cast<const uint4*>(uwrow + bbk);
        bu.q[1] = *reinterpret_cast<const uint4*>(uwrow + bbk + 8);
        gacc[it] = wmma_bf16(a.v, bg.v, gacc[it]);
        uacc[it] = wmma_bf16(a.v, bu.v, uacc[it]);
      }
    }
    __syncthreads();                                       // slab consumed
  }

  // ---- per-tile: LoRA deltas, SwiGLU, weighted combine, rw accumulation ----
  float* wiBuf = ldsWi + wave * 256;
#pragma unroll
  for (int it = 0; it < 4; ++it) {
    int i0 = ib * 256 + (it * 4 + wave) * 16;
    float acc_int[8];
#pragma unroll
    for (int j = 0; j < 8; ++j) acc_int[j] = 0.f;

    for (int e = 0; e < NE; ++e) {
      v8f gd = zero, ud = zero;
      const float* tgrow = taG + ((long)e * T_TOK + (t0 + m)) * RNK;
      const float* turow = taU + ((long)e * T_TOK + (t0 + m)) * RNK;
      const float* gBrow = gB + ((long)e * I_DIM + (i0 + n)) * RNK;
      const float* uBrow = uB + ((long)e * I_DIM + (i0 + n)) * RNK;
#pragma unroll
      for (int k0 = 0; k0 < RNK; k0 += 4) {
        v2f a2, b2;
        a2.x = tgrow[k0 + kk]; a2.y = tgrow[k0 + kk + 1];
        b2.x = gBrow[k0 + kk]; b2.y = gBrow[k0 + kk + 1];
        gd = wmma_f32(a2, b2, gd);
        a2.x = turow[k0 + kk]; a2.y = turow[k0 + kk + 1];
        b2.x = uBrow[k0 + kk]; b2.y = uBrow[k0 + kk + 1];
        ud = wmma_f32(a2, b2, ud);
      }
      C8 gc, uc, gdc, udc; gc.v = gacc[it]; uc.v = uacc[it]; gdc.v = gd; udc.v = ud;
#pragma unroll
      for (int j = 0; j < 8; ++j) {
        float g = gc.f[j] + SCALING * gdc.f[j];
        float u = uc.f[j] + SCALING * udc.f[j];
        float inter = (g / (1.f + __expf(-g))) * u;       // silu(g)*u
        float w = w_dense[(t0 + j + mh) * NE + e];
        float wi = w * inter;
        acc_int[j] += wi;
        wiBuf[(j + mh) * 16 + n] = wi;                    // same-wave LDS, in-order
      }
      // rw_e[t][r] += sum_k wi[t][k] * dA[e][r][i0+k]
      const float* dArow = dA + ((long)e * RNK + n) * I_DIM + i0;
      v8f rv = rwv[e];
#pragma unroll
      for (int k0 = 0; k0 < 16; k0 += 4) {
        v2f a2, b2;
        a2.x = wiBuf[m * 16 + k0 + kk]; a2.y = wiBuf[m * 16 + k0 + kk + 1];
        b2.x = dArow[k0 + kk];          b2.y = dArow[k0 + kk + 1];
        rv = wmma_f32(a2, b2, rv);
      }
      rwv[e] = rv;
    }
    for (int j = 0; j < 8; ++j)
      aib[(long)(t0 + j + mh) * I_DIM + i0 + n] = f2bf(acc_int[j]);
  }
  for (int e = 0; e < NE; ++e) {
    C8 rc; rc.v = rwv[e];
    for (int j = 0; j < 8; ++j)
      atomicAdd(&rw[((long)e * T_TOK + (t0 + j + mh)) * RNK + n], rc.f[j]);
  }
}

// ---------- 5) down: out = acc_int @ down_W^T + SCALING * sum_e rw_e @ dB_e^T ----------
// Block: 4 waves, 16 tokens x 128 H. acc_int tile double-buffered TDM pipeline.
__global__ __launch_bounds__(128) void k_down(
                       const unsigned short* __restrict__ aib,
                       const unsigned short* __restrict__ dWb,
                       const float* __restrict__ rw,
                       const float* __restrict__ dB,
                       float* __restrict__ out) {
  __shared__ unsigned short albs[2 * SLAB_E];  // 2 x 16KB TDM ping-pong buffers
  int tt = blockIdx.x, hb = blockIdx.y;
  int tid  = threadIdx.x;
  int wave = tid >> 5;
  int lane = tid & 31;
  int t0  = tt * 16;
  int h0a = hb * 128 + wave * 32;
  int h0b = h0a + 16;
  int n   = lane & 15;
  int m   = n;
  int abk = (lane < 16) ? 0 : 16;
  int bbk = (lane < 16) ? 0 : 16;
  int kk  = (lane < 16) ? 0 : 2;
  int mh  = (lane < 16) ? 0 : 8;
  v8f c0 = {}, c1 = {};

  unsigned int ldsBase = (unsigned int)(uintptr_t)albs;
  constexpr int NSLAB = I_DIM / SLAB;          // 16

  if (wave == 0)
    tdm_load_2d_bf16(ldsBase, aib + (long)t0 * I_DIM, SLAB, 16u,
                     (unsigned long long)I_DIM);
  for (int cI = 0; cI < NSLAB; ++cI) {
    if (wave == 0) __builtin_amdgcn_s_wait_tensorcnt(0);   // slab cI landed
    __syncthreads();
    if (wave == 0 && cI + 1 < NSLAB)                       // overlap DMA of cI+1
      tdm_load_2d_bf16(ldsBase + ((cI + 1) & 1) * (SLAB_E * 2),
                       aib + (long)t0 * I_DIM + (cI + 1) * SLAB, SLAB, 16u,
                       (unsigned long long)I_DIM);
    const unsigned short* arow = albs + (cI & 1) * SLAB_E + m * SLAB;
    long icol = (long)cI * SLAB;
    const unsigned short* brow0 = dWb + (long)(h0a + n) * I_DIM + icol;
    const unsigned short* brow1 = dWb + (long)(h0b + n) * I_DIM + icol;
    for (int i0 = 0; i0 < SLAB; i0 += 32) {
      __builtin_prefetch(brow0 + i0 + 512, 0, 1);   // stream next weight chunk
      AFrag a, b0, b1;
      a.q[0]  = *reinterpret_cast<const uint4*>(arow + i0 + abk);      // ds_load_b128
      a.q[1]  = *reinterpret_cast<const uint4*>(arow + i0 + abk + 8);
      b0.q[0] = *reinterpret_cast<const uint4*>(brow0 + i0 + bbk);
      b0.q[1] = *reinterpret_cast<const uint4*>(brow0 + i0 + bbk + 8);
      b1.q[0] = *reinterpret_cast<const uint4*>(brow1 + i0 + bbk);
      b1.q[1] = *reinterpret_cast<const uint4*>(brow1 + i0 + bbk + 8);
      c0 = wmma_bf16(a.v, b0.v, c0);
      c1 = wmma_bf16(a.v, b1.v, c1);
    }
    __syncthreads();                                       // slab consumed
  }

  for (int e = 0; e < NE; ++e) {
    const float* rrow = rw + ((long)e * T_TOK + (t0 + m)) * RNK;
    const float* bra  = dB + ((long)e * H_DIM + (h0a + n)) * RNK;
    const float* brb  = dB + ((long)e * H_DIM + (h0b + n)) * RNK;
#pragma unroll
    for (int k0 = 0; k0 < RNK; k0 += 4) {
      v2f a2, b2;
      a2.x = SCALING * rrow[k0 + kk]; a2.y = SCALING * rrow[k0 + kk + 1];
      b2.x = bra[k0 + kk];            b2.y = bra[k0 + kk + 1];
      c0 = wmma_f32(a2, b2, c0);
      b2.x = brb[k0 + kk];            b2.y = brb[k0 + kk + 1];
      c1 = wmma_f32(a2, b2, c1);
    }
  }
  C8 o0, o1; o0.v = c0; o1.v = c1;
  for (int j = 0; j < 8; ++j) {
    out[(long)(t0 + j + mh) * H_DIM + h0a + n] = o0.f[j];
    out[(long)(t0 + j + mh) * H_DIM + h0b + n] = o1.f[j];
  }
}

// ---------- 6) aux loss ----------
__global__ void k_aux(const float* __restrict__ probs_sum, const float* __restrict__ sel_cnt,
                      float* __restrict__ out_aux) {
  if (threadIdx.x == 0 && blockIdx.x == 0) {
    float s = 0.f;
    for (int e = 0; e < NE; ++e)
      s += (sel_cnt[e] / (float)T_TOK) * (probs_sum[e] / (float)T_TOK);
    out_aux[0] = (float)NE * s;
  }
}

extern "C" void kernel_launch(void* const* d_in, const int* in_sizes, int n_in,
                              void* d_out, int out_size, void* d_ws, size_t ws_size,
                              hipStream_t stream) {
  const float* x        = (const float*)d_in[0];
  const float* gate_W   = (const float*)d_in[1];
  const float* up_W     = (const float*)d_in[2];
  const float* down_W   = (const float*)d_in[3];
  const float* router_W = (const float*)d_in[4];
  const float* gate_A   = (const float*)d_in[5];
  const float* gate_B   = (const float*)d_in[6];
  const float* up_A     = (const float*)d_in[7];
  const float* up_B     = (const float*)d_in[8];
  const float* down_A   = (const float*)d_in[9];
  const float* down_B   = (const float*)d_in[10];
  float* out = (float*)d_out;

  char* wsp = (char*)d_ws;
  auto carve = [&](size_t bytes) {
    char* p = wsp;
    wsp += (bytes + 255) & ~(size_t)255;
    return p;
  };
  unsigned short* xb   = (unsigned short*)carve((size_t)T_TOK * H_DIM * 2);
  unsigned short* gWb  = (unsigned short*)carve((size_t)I_DIM * H_DIM * 2);
  unsigned short* uWb  = (unsigned short*)carve((size_t)I_DIM * H_DIM * 2);
  unsigned short* dWb  = (unsigned short*)carve((size_t)H_DIM * I_DIM * 2);
  unsigned short* aib  = (unsigned short*)carve((size_t)T_TOK * I_DIM * 2);
  float* taG     = (float*)carve((size_t)NE * T_TOK * RNK * 4);
  float* taU     = (float*)carve((size_t)NE * T_TOK * RNK * 4);
  float* rw      = (float*)carve((size_t)NE * T_TOK * RNK * 4);
  float* w_dense = (float*)carve((size_t)T_TOK * NE * 4);
  float* psum    = (float*)carve(NE * 4);
  float* scnt    = (float*)carve(NE * 4);

  k_cvt<<<2048, 256, 0, stream>>>(x,      xb,  (long)T_TOK * H_DIM);
  k_cvt<<<4096, 256, 0, stream>>>(gate_W, gWb, (long)I_DIM * H_DIM);
  k_cvt<<<4096, 256, 0, stream>>>(up_W,   uWb, (long)I_DIM * H_DIM);
  k_cvt<<<4096, 256, 0, stream>>>(down_W, dWb, (long)H_DIM * I_DIM);

  hipMemsetAsync(rw,   0, (size_t)NE * T_TOK * RNK * 4, stream);
  hipMemsetAsync(psum, 0, NE * 4, stream);
  hipMemsetAsync(scnt, 0, NE * 4, stream);

  k_router<<<T_TOK, 32, 0, stream>>>(x, router_W, w_dense, psum, scnt);
  k_loraA<<<dim3(T_TOK / 16, NE), 32, 0, stream>>>(xb, gate_A, up_A, taG, taU);
  k_main<<<dim3(T_TOK / 16, I_DIM / 256), 128, 0, stream>>>(
      xb, gWb, uWb, taG, taU, gate_B, up_B, down_A, w_dense, aib, rw);
  k_down<<<dim3(T_TOK / 16, H_DIM / 128), 128, 0, stream>>>(aib, dWb, rw, down_B, out);
  k_aux<<<1, 32, 0, stream>>>(psum, scnt, out + (size_t)T_TOK * H_DIM);
}
